// AttentionContract_10101763080616
// MI455X (gfx1250) — matrix operand order repaired and checked by reading, not compiled, not run on hardware
//
#include <hip/hip_runtime.h>

// ---------------------------------------------------------------------------
// AttentionContract for MI455X (gfx1250)
// Fused: leaky_relu -> [GEMM1 via v_wmma_f32_16x16x32_bf16 + LayerNorm] and
//        [GEMM2 logits -> mask(0-fill) -> softmax over Y] -> weighted
//        contraction over Y.
// Single HBM pass over xy_feat: the full 256x128 lr tile for one x is staged
// as bf16 in LDS (64KB, XOR-swizzled 16B chunks for conflict-free b128 reads)
// while the W2 logits are computed in the same pass.
// ---------------------------------------------------------------------------

typedef __attribute__((ext_vector_type(16))) __bf16 v16bf;
typedef __attribute__((ext_vector_type(8)))  float  v8f;

union FragBF16 {
  v16bf v;
  unsigned short u[16];
  uint4 q[2];
};

__device__ __forceinline__ unsigned short f2bf(float f) {
  // round-to-nearest-even f32 -> bf16 (bit-level)
  unsigned int u = __float_as_uint(f);
  unsigned int r = u + 0x7FFFu + ((u >> 16) & 1u);
  return (unsigned short)(r >> 16);
}
__device__ __forceinline__ float lrelu(float v) { return v > 0.0f ? v : 0.01f * v; }

#define XDIM 2048
#define YDIM 256
#define FDIM 128
#define HN   256   // H * NF
#define HEADS 8

__global__ __launch_bounds__(256) void attn_contract_kernel(
    const float* __restrict__ xy_feat,   // (X, Y, F)
    const int*   __restrict__ mask,      // (X, Y)
    const float* __restrict__ W1,        // (F, HN) row-major
    const float* __restrict__ b1,        // (HN)
    const float* __restrict__ gamma_,    // (HN)
    const float* __restrict__ beta_,     // (HN)
    const float* __restrict__ W2,        // (F, H) row-major
    const float* __restrict__ b2,        // (H)
    float*       __restrict__ out)       // (X, HN)
{
  // Full bf16 lr tile for this x: 256 rows x 128 cols = 64KB.
  // Each row is 16 chunks of 16B; chunk c of row r stored at c ^ (r & 15)
  // so half-wave b128 reads at fixed column across 16 rows hit distinct banks.
  __shared__ __align__(16) unsigned short s_lr[YDIM * FDIM];  // 64KB
  __shared__ float s_hid[16 * HN];                            // 16KB
  __shared__ float s_coef[YDIM * HEADS];                      // 8KB
  __shared__ float s_mu[16];
  __shared__ float s_rstd[16];

  const int t    = threadIdx.x;      // 0..255
  const int x    = blockIdx.x;       // 0..2047
  const int lane = t & 31;           // wave32 lane
  const int w    = t >> 5;           // wave id 0..7
  const int nlo  = lane & 15;
  const int hi   = lane >> 4;        // lane-half (K-offset selector, ISA layout)
  const int seg  = t & 15;           // F-segment owner (8 floats each)
  const int rgrp = t >> 4;           // row group base

  // ======== Phase A: single pass over xy_feat ===============================
  // Stage lr (bf16, swizzled) into LDS AND accumulate W2 logits per row.
  // Thread t owns F-columns [seg*8, seg*8+8); its W2 slice lives in registers.
  {
    float w2r[8 * HEADS];
#pragma unroll
    for (int j = 0; j < 8; ++j)
#pragma unroll
      for (int h = 0; h < HEADS; ++h)
        w2r[j * HEADS + h] = W2[(seg * 8 + j) * HEADS + h];
    float b2r[HEADS];
#pragma unroll
    for (int h = 0; h < HEADS; ++h) b2r[h] = b2[h];

    for (int i = 0; i < 16; ++i) {
      const int row = rgrp + 16 * i;  // y index
      const float* src = xy_feat + ((size_t)x * YDIM + row) * FDIM + seg * 8;
      float4 a = ((const float4*)src)[0];
      float4 b = ((const float4*)src)[1];
      float l[8] = {lrelu(a.x), lrelu(a.y), lrelu(a.z), lrelu(a.w),
                    lrelu(b.x), lrelu(b.y), lrelu(b.z), lrelu(b.w)};
      // swizzled bf16 store: chunk index = seg, physical = seg ^ (row & 15)
      unsigned short* d = &s_lr[row * FDIM + (seg ^ (row & 15)) * 8];
#pragma unroll
      for (int j = 0; j < 8; ++j) d[j] = f2bf(l[j]);
      if (i + 1 < 16) __builtin_prefetch(src + 16 * FDIM, 0, 0);  // global_prefetch_b8

      // partial logits for this row over owned F-segment
      float lg[HEADS];
#pragma unroll
      for (int h = 0; h < HEADS; ++h) {
        float s = 0.0f;
#pragma unroll
        for (int j = 0; j < 8; ++j) s += l[j] * w2r[j * HEADS + h];
        lg[h] = s;
      }
      // reduce across the 16 segment-owners of this row (consecutive lanes)
#pragma unroll
      for (int off = 8; off; off >>= 1)
#pragma unroll
        for (int h = 0; h < HEADS; ++h) lg[h] += __shfl_xor(lg[h], off, 16);
      if (seg == 0) {
        const bool msk = mask[(size_t)x * YDIM + row] != 0;
#pragma unroll
        for (int h = 0; h < HEADS; ++h)
          s_coef[row * HEADS + h] = msk ? 0.0f : (lg[h] + b2r[h]);
      }
    }
  }
  __syncthreads();

  // ======== Phase B: softmax over Y per head; wave w owns head w ============
  if (w < HEADS) {
    const int h = w;
    float vals[8];
    float m = -3.4e38f;
#pragma unroll
    for (int i = 0; i < 8; ++i) {
      vals[i] = s_coef[(lane + 32 * i) * HEADS + h];
      m = fmaxf(m, vals[i]);
    }
#pragma unroll
    for (int off = 16; off; off >>= 1) m = fmaxf(m, __shfl_xor(m, off, 32));
    float s = 0.0f;
#pragma unroll
    for (int i = 0; i < 8; ++i) { vals[i] = __expf(vals[i] - m); s += vals[i]; }
#pragma unroll
    for (int off = 16; off; off >>= 1) s += __shfl_xor(s, off, 32);
    const float inv = 1.0f / s;
#pragma unroll
    for (int i = 0; i < 8; ++i) s_coef[(lane + 32 * i) * HEADS + h] = vals[i] * inv;
  }

  // ======== Phase C: WMMA GEMM1 + LayerNorm + weighted Y-contraction ========
  // B-fragments (W1 bf16) in registers for the whole block.
  // Wave w owns N-tiles {2w, 2w+1}; 4 K-steps of 32 cover F=128.
  // B layout used here (contiguous K, same as A): lane = N in tile, lane-half selects K block,
  // v16bf element j -> K = kt*32 + hi*16 + j.
  FragBF16 Bf[2][4];
#pragma unroll
  for (int tl = 0; tl < 2; ++tl) {
    const int n = (2 * w + tl) * 16 + nlo;
#pragma unroll
    for (int kt = 0; kt < 4; ++kt) {
#pragma unroll
      for (int j = 0; j < 16; ++j) {
        const int K = kt * 32 + hi * 16 + j;
        Bf[tl][kt].u[j] = f2bf(W1[K * HN + n]);
      }
    }
  }

  const float g    = gamma_[t];
  const float bt   = beta_[t];
  const float b1c0 = b1[(2 * w + 0) * 16 + nlo];
  const float b1c1 = b1[(2 * w + 1) * 16 + nlo];
  float acc = 0.0f;  // output column t, accumulated over all Y

  __syncthreads();  // s_lr fully staged, s_coef softmaxed

  for (int yc = 0; yc < 16; ++yc) {
    // ---- 8 WMMAs: 2 N-tiles x 4 K-steps, bf16 -> f32 accumulate ----
    v8f c0 = {};
    v8f c1 = {};
    const int arow = yc * 16 + nlo;  // A row; arow & 15 == nlo
#pragma unroll
    for (int kt = 0; kt < 4; ++kt) {
      FragBF16 A;
      // A layout (matches B): elem j -> K = kt*32 + hi*16 + j.
      // 16B chunk indices kt*4+2*hi and kt*4+2*hi+1, XOR-deswizzled by row.
      const int c0i = (kt * 4 + 2 * hi) ^ nlo;
      const int c1i = (kt * 4 + 2 * hi + 1) ^ nlo;
      A.q[0] = *(const uint4*)&s_lr[arow * FDIM + c0i * 8];
      A.q[1] = *(const uint4*)&s_lr[arow * FDIM + c1i * 8];
      c0 = __builtin_amdgcn_wmma_f32_16x16x32_bf16(false, A.v, false, Bf[0][kt].v,
                                                   (short)0, c0, false, false);
      c1 = __builtin_amdgcn_wmma_f32_16x16x32_bf16(false, A.v, false, Bf[1][kt].v,
                                                   (short)0, c1, false, false);
    }

    // ---- spill hid (+b1) to LDS: VGPR r -> row hi*8+r, col = tile*16 + nlo ----
    {
      const int col0 = (2 * w + 0) * 16 + nlo;
      const int col1 = (2 * w + 1) * 16 + nlo;
      const int rb   = hi * 8;
#pragma unroll
      for (int r = 0; r < 8; ++r) {
        s_hid[(rb + r) * HN + col0] = c0[r] + b1c0;
        s_hid[(rb + r) * HN + col1] = c1[r] + b1c1;
      }
    }
    __syncthreads();

    // ---- per-row LayerNorm stats: 16 threads/row, width-16 shuffle reduce ----
    {
      const float* hp = &s_hid[rgrp * HN + seg * 16];
      float s = 0.0f, s2 = 0.0f;
#pragma unroll
      for (int i = 0; i < 16; ++i) { float v = hp[i]; s += v; s2 += v * v; }
#pragma unroll
      for (int off = 8; off; off >>= 1) {
        s  += __shfl_xor(s, off, 16);
        s2 += __shfl_xor(s2, off, 16);
      }
      if (seg == 0) {
        const float mu  = s * (1.0f / HN);
        const float var = s2 * (1.0f / HN) - mu * mu;
        s_mu[rgrp]   = mu;
        s_rstd[rgrp] = rsqrtf(var + 1e-5f);
      }
    }
    __syncthreads();

    // ---- weighted contraction over this chunk's 16 rows; thread t = col t ----
    {
      const int h = t >> 5;  // NF = 32 -> head = n/32
#pragma unroll
      for (int mm = 0; mm < 16; ++mm) {
        const float cf = s_coef[(yc * 16 + mm) * HEADS + h];
        const float v  = (s_hid[mm * HN + t] - s_mu[mm]) * s_rstd[mm] * g + bt;
        acc += cf * v;
      }
    }
    __syncthreads();  // protect s_hid before next chunk
  }

  out[(size_t)x * HN + t] = acc;
}

extern "C" void kernel_launch(void* const* d_in, const int* in_sizes, int n_in,
                              void* d_out, int out_size, void* d_ws, size_t ws_size,
                              hipStream_t stream) {
  (void)in_sizes; (void)n_in; (void)out_size; (void)d_ws; (void)ws_size;
  const float* xy    = (const float*)d_in[0];
  const int*   mask  = (const int*)  d_in[1];
  const float* W1    = (const float*)d_in[2];
  const float* b1    = (const float*)d_in[3];
  const float* gam   = (const float*)d_in[4];
  const float* bet   = (const float*)d_in[5];
  const float* W2    = (const float*)d_in[6];
  const float* b2    = (const float*)d_in[7];
  float* out = (float*)d_out;
  attn_contract_kernel<<<XDIM, 256, 0, stream>>>(xy, mask, W1, b1, gam, bet, W2, b2, out);
}
